// MixtureOfExperts_21457656610886
// MI455X (gfx1250) — matrix operand order repaired and checked by reading, not compiled, not run on hardware
//
#include <hip/hip_runtime.h>
#include <hip/hip_bf16.h>
#include <math.h>

// ---------------- problem constants ----------------
constexpr int N = 16384;
constexpr int H = 1024;
constexpr int C = 256;
constexpr int E = 16;
constexpr int R = 128;
constexpr float EPS = 1e-8f;
constexpr int TM = 64;               // expert kernel M-tile (tokens per block)

// ---------------- WMMA types ----------------
typedef __attribute__((ext_vector_type(16))) __bf16 bf16x16;
typedef __attribute__((ext_vector_type(8)))  __bf16 bf16x8;
typedef __attribute__((ext_vector_type(4)))  __bf16 bf16x4;
typedef __attribute__((ext_vector_type(8)))  float  f32x8;

__device__ inline f32x8 wmma_bf16(bf16x16 a, bf16x16 b, f32x8 c) {
    return __builtin_amdgcn_wmma_f32_16x16x32_bf16(
        /*neg_a=*/false, a, /*neg_b=*/false, b,
        /*c_mod=*/(short)0, c, /*reuse_a=*/false, /*reuse_b=*/false);
}

// A-fragment (16x32 bf16) from row-major LDS row pointer.
// Lane l (row = l%16): K = k0 + (l>=16?16:0) .. +7   in VGPR 0..3
//                      K = k0+8 + (l>=16?16:0) .. +7 in VGPR 4..7
__device__ inline bf16x16 load_a_frag(const __bf16* row, int k0, int lhalf) {
    union { bf16x16 v; bf16x8 h[2]; } u;
    u.h[0] = *(const bf16x8*)(row + k0 + lhalf * 16);
    u.h[1] = *(const bf16x8*)(row + k0 + 8 + lhalf * 16);
    return u.v;
}

// ---------------- workspace layout (bytes) ----------------
constexpr size_t WD_OFF   = 0;                        // 1024*256*2     = 512 KB
constexpr size_t WG_OFF   = WD_OFF + 524288;          // 16*1024*1024*2 = 32 MB
constexpr size_t U_OFF    = WG_OFF + 33554432;        // 16*1024*128*2  = 4 MB
constexpr size_t V_OFF    = U_OFF  + 4194304;         // 16*128*1024*2  = 4 MB
constexpr size_t XBF_OFF  = V_OFF  + 4194304;         // N*H*2          = 32 MB
constexpr size_t DIST_OFF = XBF_OFF + 33554432;       // N*C*4          = 16 MB
constexpr size_t CNT_OFF  = DIST_OFF + 16777216;      // 256 B
constexpr size_t TL_OFF   = CNT_OFF + 256;            // E*N*4          = 1 MB
constexpr size_t WL_OFF   = TL_OFF + 1048576;         // E*N*4          = 1 MB

// ---------------- kernel: zero output + counts ----------------
__global__ void zero_kernel(float* out, int* counts) {
    size_t gid = (size_t)blockIdx.x * blockDim.x + threadIdx.x;
    size_t total = (size_t)N * H;
    for (size_t i = gid; i < total; i += (size_t)gridDim.x * blockDim.x)
        out[i] = 0.0f;
    if (gid < E) counts[gid] = 0;
}

// ---------------- kernel: x (f32, row-major) -> bf16 copy -------------------
__global__ void pack_x_bf16_kernel(const float* __restrict__ x,
                                   __bf16* __restrict__ xbf) {
    size_t i = (size_t)blockIdx.x * blockDim.x + threadIdx.x;
    size_t base = i * 4;
    if (base >= (size_t)N * H) return;
    float4 v = *(const float4*)(x + base);
    bf16x4 b;
    b[0] = (__bf16)v.x; b[1] = (__bf16)v.y;
    b[2] = (__bf16)v.z; b[3] = (__bf16)v.w;
    *(bf16x4*)(xbf + base) = b;
}

// ---------------- kernel: pack f32 [Eg][K][Nc] -> bf16 B-fragment layout ----
// dst[((e*Nt + t)*Kb + kb)*256 + c*16 + kk] = src[e][kb*16+kk][t*16+c]
__global__ void pack_bfrag_kernel(const float* __restrict__ src,
                                  __bf16* __restrict__ dst,
                                  int K, int Nc, size_t total) {
    size_t gid = (size_t)blockIdx.x * blockDim.x + threadIdx.x;
    if (gid >= total) return;
    int Kb = K >> 4;
    int Nt = Nc >> 4;
    int kk = gid & 15;
    int c  = (gid >> 4) & 15;
    size_t rest = gid >> 8;
    int kb = (int)(rest % Kb);
    size_t rest2 = rest / Kb;
    int t = (int)(rest2 % Nt);
    int e = (int)(rest2 / Nt);
    size_t sidx = ((size_t)e * K + (size_t)kb * 16 + kk) * Nc + (size_t)t * 16 + c;
    dst[gid] = (__bf16)src[sidx];
}

// ---------------- staging: gathered bf16 rows -> LDS via async DMA ----------
// Each 16B chunk moved memory->LDS with GLOBAL_LOAD_ASYNC_TO_LDS_B128
// (ASYNCcnt-tracked, no VGPR data path). Caller barriers after return.
template <int ROWS>
__device__ inline void stage_rows_async(const __bf16* __restrict__ xbf,
                                        const int* tok_s, __bf16* Xs, int tid) {
    constexpr int CH_PER_ROW = (H * 2) / 16;          // 128 x 16B per row
    constexpr int TOTAL = ROWS * CH_PER_ROW;
    #pragma unroll 4
    for (int i = tid; i < TOTAL; i += 256) {
        int row = i >> 7;            // / CH_PER_ROW
        int c   = i & (CH_PER_ROW - 1);
        // low 32 bits of a flat shared pointer == LDS byte offset (aperture rule)
        uint32_t lds_off = (uint32_t)(uintptr_t)(&Xs[row * H + c * 8]);
        uint64_t gaddr   = (uint64_t)(uintptr_t)(xbf + (size_t)tok_s[row] * H + c * 8);
        asm volatile("global_load_async_to_lds_b128 %0, %1, off"
                     :: "v"(lds_off), "v"(gaddr) : "memory");
    }
    asm volatile("s_wait_asynccnt 0x0" ::: "memory");
}

// ---------------- kernel: distilled = gelu(x @ Wd + bd) via WMMA ------------
__global__ __launch_bounds__(256) void router_gemm_kernel(
    const __bf16* __restrict__ xbf, const __bf16* __restrict__ WdP,
    const float* __restrict__ bd, float* __restrict__ distilled) {
    __shared__ alignas(16) __bf16 Xs[16 * H];
    __shared__ int tok_s[16];
    const int tid = threadIdx.x;
    const int t0 = blockIdx.x * 16;
    if (tid < 16) tok_s[tid] = t0 + tid;
    __syncthreads();
    stage_rows_async<16>(xbf, tok_s, Xs, tid);
    __syncthreads();

    const int wave = tid >> 5, lane = tid & 31;
    const int lhalf = lane >> 4, lmod = lane & 15;
    constexpr int Kb = H / 16;   // 64

    #pragma unroll
    for (int j = 0; j < 2; ++j) {               // C/16 = 16 tiles, 8 waves
        const int nt = wave * 2 + j;
        f32x8 acc = {};
        const __bf16* Bb = WdP + (size_t)nt * Kb * 256;
        for (int k0 = 0; k0 < H; k0 += 32) {
            bf16x16 a = load_a_frag(&Xs[lmod * H], k0, lhalf);
            bf16x16 b = *(const bf16x16*)(Bb + ((k0 >> 4) + lhalf) * 256 + lmod * 16);
            acc = wmma_bf16(a, b, acc);
        }
        const int n = nt * 16 + lmod;
        const float bdn = bd[n];
        #pragma unroll
        for (int v = 0; v < 8; ++v) {
            int m = v + 8 * lhalf;
            float s = acc[v] + bdn;
            float g = 0.5f * s * (1.0f + erff(s * 0.70710678118654752f));
            distilled[(size_t)(t0 + m) * C + n] = g;
        }
    }
}

// ---------------- kernel: distances, softmax, top-2, expert lists -----------
__global__ __launch_bounds__(256) void router_topk_kernel(
    const float* __restrict__ distilled, const float* __restrict__ centroids,
    int* __restrict__ counts, int* __restrict__ tlist, float* __restrict__ wlist) {
    __shared__ float cs[E * C];
    __shared__ float cninv[E];
    __shared__ float cn2[E];
    const int tid = threadIdx.x;
    for (int i = tid; i < E * C; i += 256) cs[i] = centroids[i];
    __syncthreads();
    if (tid < E) {
        float s = 0.0f;
        for (int c = 0; c < C; ++c) { float v = cs[tid * C + c]; s += v * v; }
        float inv = 1.0f / fmaxf(sqrtf(s), EPS);
        cninv[tid] = inv;
        cn2[tid] = s * inv * inv;
    }
    __syncthreads();

    const int token = blockIdx.x * 256 + tid;
    const float* d = distilled + (size_t)token * C;
    float ss = 0.0f;
    float dot[E];
    #pragma unroll
    for (int e = 0; e < E; ++e) dot[e] = 0.0f;
    for (int c = 0; c < C; ++c) {
        float dv = d[c];
        ss += dv * dv;
        #pragma unroll
        for (int e = 0; e < E; ++e) dot[e] += dv * cs[e * C + c];
    }
    float ind = 1.0f / fmaxf(sqrtf(ss), EPS);
    float sdn = ss * ind * ind;

    float sc[E]; float mx = -1e30f;
    #pragma unroll
    for (int e = 0; e < E; ++e) {
        float dnc = dot[e] * ind * cninv[e];
        float d2 = sdn + cn2[e] - 2.0f * dnc;
        float dist = sqrtf(fmaxf(d2, 0.0f));
        sc[e] = -dist;
        mx = fmaxf(mx, sc[e]);
    }
    float sum = 0.0f; float p[E];
    #pragma unroll
    for (int e = 0; e < E; ++e) { p[e] = __expf(sc[e] - mx); sum += p[e]; }
    float inv_sum = 1.0f / sum;
    #pragma unroll
    for (int e = 0; e < E; ++e) p[e] *= inv_sum;

    // top-2 (first index on ties, matching lax.top_k)
    int i1 = 0; float p1 = p[0];
    #pragma unroll
    for (int e = 1; e < E; ++e) if (p[e] > p1) { p1 = p[e]; i1 = e; }
    int i2 = -1; float p2 = -1.0f;
    #pragma unroll
    for (int e = 0; e < E; ++e) {
        if (e == i1) continue;
        if (p[e] > p2) { p2 = p[e]; i2 = e; }
    }
    int pos1 = atomicAdd(&counts[i1], 1);
    tlist[(size_t)i1 * N + pos1] = token;
    wlist[(size_t)i1 * N + pos1] = p1;
    int pos2 = atomicAdd(&counts[i2], 1);
    tlist[(size_t)i2 * N + pos2] = token;
    wlist[(size_t)i2 * N + pos2] = p2;
}

// ---------------- kernel: grouped expert highway blocks ---------------------
// grid = E * (N/TM) blocks; block handles TM=64 gathered tokens of one expert.
// 8 waves; each wave owns 128 output columns and all 4 M-tiles (register-level
// B-fragment reuse x4 -> 4x less L2 traffic on Wg/U/V).
__global__ __launch_bounds__(256) void expert_kernel(
    const float* __restrict__ x, const __bf16* __restrict__ xbf,
    const float* __restrict__ bg,
    const __bf16* __restrict__ WgP, const __bf16* __restrict__ UP,
    const __bf16* __restrict__ VP,
    const int* __restrict__ counts, const int* __restrict__ tlist,
    const float* __restrict__ wlist, float* __restrict__ out) {
    __shared__ alignas(16) __bf16 Xs[TM * H];      // 128 KB
    __shared__ alignas(16) __bf16 T1s[TM * R];     // 16 KB
    __shared__ int   tok_s[TM];
    __shared__ float w_s[TM];

    const int e    = blockIdx.x >> 8;      // / (N/TM=256)
    const int tile = blockIdx.x & 255;
    const int cnt  = counts[e];
    if (tile * TM >= cnt) return;

    const int tid = threadIdx.x;
    if (tid < TM) {
        int idx = tile * TM + tid;
        bool valid = idx < cnt;
        tok_s[tid] = tlist[(size_t)e * N + (valid ? idx : tile * TM)];
        w_s[tid]   = valid ? wlist[(size_t)e * N + idx] : 0.0f;
    }
    __syncthreads();

    stage_rows_async<TM>(xbf, tok_s, Xs, tid);
    __syncthreads();

    const int wave = tid >> 5, lane = tid & 31;
    const int lhalf = lane >> 4, lmod = lane & 15;
    constexpr int KbH = H / 16;   // 64
    constexpr int KbR = R / 16;   // 8
    constexpr int NtH = H / 16;   // 64
    constexpr int NtR = R / 16;   // 8
    constexpr int MT  = TM / 16;  // 4 M-tiles

    // ---- phase B: t1 = X @ U_e  (wave w -> n-tile w of R=128, all M-tiles) --
    {
        f32x8 acc[MT];
        #pragma unroll
        for (int mt = 0; mt < MT; ++mt) acc[mt] = (f32x8){};
        const __bf16* Bb = UP + ((size_t)e * NtR + wave) * KbH * 256;
        for (int k0 = 0; k0 < H; k0 += 32) {
            bf16x16 b = *(const bf16x16*)(Bb + ((k0 >> 4) + lhalf) * 256 + lmod * 16);
            #pragma unroll
            for (int mt = 0; mt < MT; ++mt) {
                bf16x16 a = load_a_frag(&Xs[(mt * 16 + lmod) * H], k0, lhalf);
                acc[mt] = wmma_bf16(a, b, acc[mt]);
            }
        }
        #pragma unroll
        for (int mt = 0; mt < MT; ++mt)
            #pragma unroll
            for (int v = 0; v < 8; ++v) {
                int m = mt * 16 + v + 8 * lhalf;
                T1s[m * R + wave * 16 + lmod] = (__bf16)acc[mt][v];
            }
    }
    __syncthreads();

    // ---- phase C: per wave 128 output columns, all 4 M-tiles ----
    for (int j = 0; j < 8; ++j) {
        const int nt = wave * 8 + j;
        const int n  = nt * 16 + lmod;

        f32x8 g[MT];
        #pragma unroll
        for (int mt = 0; mt < MT; ++mt) g[mt] = (f32x8){};
        const __bf16* Wb = WgP + ((size_t)e * NtH + nt) * KbH * 256;
        for (int k0 = 0; k0 < H; k0 += 32) {
            bf16x16 b = *(const bf16x16*)(Wb + ((k0 >> 4) + lhalf) * 256 + lmod * 16);
            #pragma unroll
            for (int mt = 0; mt < MT; ++mt) {
                bf16x16 a = load_a_frag(&Xs[(mt * 16 + lmod) * H], k0, lhalf);
                g[mt] = wmma_bf16(a, b, g[mt]);
            }
        }

        f32x8 t[MT];
        #pragma unroll
        for (int mt = 0; mt < MT; ++mt) t[mt] = (f32x8){};
        const __bf16* Vb = VP + ((size_t)e * NtH + nt) * KbR * 256;
        #pragma unroll
        for (int k0 = 0; k0 < R; k0 += 32) {
            bf16x16 b = *(const bf16x16*)(Vb + ((k0 >> 4) + lhalf) * 256 + lmod * 16);
            #pragma unroll
            for (int mt = 0; mt < MT; ++mt) {
                bf16x16 a = load_a_frag(&T1s[(mt * 16 + lmod) * R], k0, lhalf);
                t[mt] = wmma_bf16(a, b, t[mt]);
            }
        }

        const float bgn = bg[(size_t)e * H + n];
        #pragma unroll
        for (int mt = 0; mt < MT; ++mt) {
            #pragma unroll
            for (int v = 0; v < 8; ++v) {
                int m = mt * 16 + v + 8 * lhalf;
                float wm = w_s[m];
                if (wm != 0.0f) {
                    size_t row = (size_t)tok_s[m] * H;
                    float gate = 1.0f / (1.0f + __expf(-(g[mt][v] + bgn)));
                    float xv = x[row + n];
                    float y = gate * t[mt][v] + (1.0f - gate) * xv;
                    atomicAdd(out + row + n, wm * y);
                }
            }
        }
    }
}

// ---------------- host launcher ----------------
extern "C" void kernel_launch(void* const* d_in, const int* in_sizes, int n_in,
                              void* d_out, int out_size, void* d_ws, size_t ws_size,
                              hipStream_t stream) {
    const float* x    = (const float*)d_in[0];
    const float* Wd   = (const float*)d_in[1];
    const float* bd   = (const float*)d_in[2];
    const float* cent = (const float*)d_in[3];
    const float* Wg   = (const float*)d_in[4];
    const float* bg   = (const float*)d_in[5];
    const float* U    = (const float*)d_in[6];
    const float* V    = (const float*)d_in[7];
    (void)in_sizes; (void)n_in; (void)out_size; (void)ws_size;

    char* ws = (char*)d_ws;
    __bf16* WdP = (__bf16*)(ws + WD_OFF);
    __bf16* WgP = (__bf16*)(ws + WG_OFF);
    __bf16* UP  = (__bf16*)(ws + U_OFF);
    __bf16* VP  = (__bf16*)(ws + V_OFF);
    __bf16* xbf = (__bf16*)(ws + XBF_OFF);
    float* dist = (float*)(ws + DIST_OFF);
    int*   cnts = (int*)(ws + CNT_OFF);
    int*   tl   = (int*)(ws + TL_OFF);
    float* wl   = (float*)(ws + WL_OFF);
    float* out  = (float*)d_out;

    // 1) zero out + counts
    zero_kernel<<<4096, 256, 0, stream>>>(out, cnts);

    // 2) pack x + weights to bf16 (weights in B-fragment layout)
    pack_x_bf16_kernel<<<(N * H / 4 + 255) / 256, 256, 0, stream>>>(x, xbf);
    {
        size_t tWd = (size_t)H * C;
        pack_bfrag_kernel<<<(unsigned)((tWd + 255) / 256), 256, 0, stream>>>(Wd, WdP, H, C, tWd);
        size_t tWg = (size_t)E * H * H;
        pack_bfrag_kernel<<<(unsigned)((tWg + 255) / 256), 256, 0, stream>>>(Wg, WgP, H, H, tWg);
        size_t tU = (size_t)E * H * R;
        pack_bfrag_kernel<<<(unsigned)((tU + 255) / 256), 256, 0, stream>>>(U, UP, H, R, tU);
        size_t tV = (size_t)E * R * H;
        pack_bfrag_kernel<<<(unsigned)((tV + 255) / 256), 256, 0, stream>>>(V, VP, R, H, tV);
    }

    // 3) router GEMM (distilled)
    router_gemm_kernel<<<N / 16, 256, 0, stream>>>(xbf, WdP, bd, dist);

    // 4) distances + softmax + top-2 + expert lists
    router_topk_kernel<<<N / 256, 256, 0, stream>>>(dist, cent, cnts, tl, wl);

    // 5) grouped expert compute (worst-case capacity grid, early exit)
    expert_kernel<<<E * (N / TM), 256, 0, stream>>>(x, xbf, bg, WgP, UP, VP, cnts, tl, wl, out);
}
